// PointExpandVoxel3_80006650790057
// MI455X (gfx1250) — matrix operand order repaired and checked by reading, compiled, not run on hardware
//
#include <hip/hip_runtime.h>
#include <hip/hip_bf16.h>
#include <math.h>

#define BB 4
#define HH 200
#define WW 176
#define CC 256
#define NN 16384
#define PP 65536
#define SS (BB*HH*WW)      /* 140800 */
#define HP (HH+2)
#define WP (WW+2)
#define BN_EPS 1e-3f
#define RPB 128            /* rows per block in column-stats */

typedef unsigned short u16;
typedef unsigned int   u32;
typedef __attribute__((ext_vector_type(16))) __bf16 v16bf;
typedef __attribute__((ext_vector_type(8)))  float  v8f;

union ABFrag { v16bf v; u32 u[8]; u16 s[16]; };

__device__ __forceinline__ u16 f2bf(float f) {
    u32 u = __float_as_uint(f);
    u32 r = (u + 0x7FFFu + ((u >> 16) & 1u)) >> 16;   // RNE
    return (u16)r;
}

// CDNA5 async global->LDS path (ASYNCcnt), guarded so either toolchain compiles.
#if __has_builtin(__builtin_amdgcn_global_load_async_to_lds_b128) && \
    __has_builtin(__builtin_amdgcn_s_wait_asynccnt)
#define HAS_ASYNC_LDS 1
typedef int v4i_ld __attribute__((vector_size(4 * sizeof(int))));
typedef __attribute__((address_space(1))) v4i_ld as1_v4i;
typedef __attribute__((address_space(3))) v4i_ld as3_v4i;
__device__ __forceinline__ as1_v4i* to_as1(const void* p) {
    return (as1_v4i*)(unsigned long long)(size_t)p;
}
__device__ __forceinline__ as3_v4i* to_as3(void* p) {
    // generic LDS pointer: low 32 bits are the wave-relative LDS byte address
    return (as3_v4i*)(unsigned int)(size_t)p;
}
#else
#define HAS_ASYNC_LDS 0
#endif

// ---------------------------------------------------------------------------
// Stage 1: [B,C,N] -> [B*N, C] transpose through LDS (coalesced both sides)
// ---------------------------------------------------------------------------
__global__ void k_transpose(const float* __restrict__ feats, float* __restrict__ pw) {
    __shared__ float tile[32][33];
    int n0 = blockIdx.x * 32, c0 = blockIdx.y * 32, b = blockIdx.z;
    int tx = threadIdx.x, ty = threadIdx.y;               // 32 x 8
    #pragma unroll
    for (int j = 0; j < 4; j++)
        tile[ty + j * 8][tx] = feats[((size_t)(b * CC + c0 + ty + j * 8)) * NN + n0 + tx];
    __syncthreads();
    #pragma unroll
    for (int j = 0; j < 4; j++)
        pw[((size_t)(b * NN + n0 + ty + j * 8)) * CC + c0 + tx] = tile[tx][ty + j * 8];
}

// ---------------------------------------------------------------------------
// Stage 2: voxel-mean pooling via float atomics
// ---------------------------------------------------------------------------
__global__ void k_scatter(const float* __restrict__ pw, const int* __restrict__ cb,
                          const int* __restrict__ cy, const int* __restrict__ cx,
                          float* __restrict__ sums, float* __restrict__ cnt) {
    int p = blockIdx.x, c = threadIdx.x;
    int vid = cb[p] * (HH * WW) + cy[p] * WW + cx[p];
    atomicAdd(&sums[(size_t)vid * CC + c], pw[(size_t)p * CC + c]);
    if (c == 0) atomicAdd(&cnt[vid], 1.0f);
}

__global__ void k_finalize(float* __restrict__ sums, const float* __restrict__ cnt,
                           u16* __restrict__ pooled_bf) {
    int s = blockIdx.x, c = threadIdx.x;
    float inv = 1.0f / fmaxf(cnt[s], 1.0f);
    size_t idx = (size_t)s * CC + c;
    float v = sums[idx] * inv;
    sums[idx] = v;                 // pooled f32 in place
    pooled_bf[idx] = f2bf(v);      // bf16 copy for WMMA
}

// ---------------------------------------------------------------------------
// Weight convert+transpose: f32 [K, Ncols] -> bf16 [Nout, K] (zero-pad rows)
// ---------------------------------------------------------------------------
__global__ void k_wtrans(const float* __restrict__ in, u16* __restrict__ out, int K, int Ncols) {
    int n = blockIdx.x;
    int k = blockIdx.y * 256 + threadIdx.x;
    if (k < K) out[(size_t)n * K + k] = (n < Ncols) ? f2bf(in[(size_t)k * Ncols + n]) : (u16)0;
}

// ---------------------------------------------------------------------------
// bf16 WMMA GEMM: D[S,Nout] = A[S,K] @ W^T[Nout,K]^T, f32 accumulate.
// 8 waves/block; WN col-tiles x (8/WN) row-tiles of 16x16.
// ---------------------------------------------------------------------------
__global__ void k_gemm(const u16* __restrict__ A, const u16* __restrict__ WT,
                       float* __restrict__ D, int K, int Nout, int WN) {
    int tid = threadIdx.x;
    int wave = tid >> 5, lane = tid & 31, half = lane >> 4, mn = lane & 15;
    int wn = wave % WN, wm = wave / WN;
    int row0 = (blockIdx.y * (8 / WN) + wm) * 16;
    int col0 = (blockIdx.x * WN + wn) * 16;
    const u32* ap = (const u32*)(A + (size_t)(row0 + mn) * K);
    const u32* bp = (const u32*)(WT + (size_t)(col0 + mn) * K);
    __builtin_prefetch(bp, 0, 3);
    v8f acc = {};
    for (int k0 = 0; k0 < K; k0 += 32) {
        ABFrag a, bf;
        int au = (k0 >> 1) + half * 8;     // A: k = k0 + half*16 + i
        #pragma unroll
        for (int j = 0; j < 8; j++) a.u[j] = ap[au + j];
        int bu = (k0 >> 1) + half * 8;     // B: k = k0 + half*16 + i
        #pragma unroll
        for (int j = 0; j < 8; j++) bf.u[j] = bp[bu + j];
        acc = __builtin_amdgcn_wmma_f32_16x16x32_bf16(false, a.v, false, bf.v,
                                                      (short)0, acc, false, false);
    }
    float* drow = D + (size_t)(row0 + half * 8) * Nout + col0 + mn;
    #pragma unroll
    for (int r = 0; r < 8; r++) drow[(size_t)r * Nout] = acc[r];
}

// ---------------------------------------------------------------------------
// BatchNorm column statistics (sum, sumsq) via per-block partials + atomics
// ---------------------------------------------------------------------------
__global__ void k_colstats(const float* __restrict__ X, float* __restrict__ csum,
                           float* __restrict__ csq, int ncols) {
    int c = threadIdx.x;
    if (c >= ncols) return;
    int r0 = blockIdx.x * RPB;
    float s = 0.f, q = 0.f;
    for (int r = 0; r < RPB; r++) {
        float v = X[(size_t)(r0 + r) * ncols + c];
        s += v; q += v * v;
    }
    atomicAdd(&csum[c], s);
    atomicAdd(&csq[c], q);
}

__global__ void k_bnrelu_bf(const float* __restrict__ X, const float* __restrict__ csum,
                            const float* __restrict__ csq, const float* __restrict__ g,
                            const float* __restrict__ b, u16* __restrict__ out, int ncols) {
    size_t idx = (size_t)blockIdx.x * 256 + threadIdx.x;
    if (idx >= (size_t)SS * ncols) return;
    int c = (int)(idx % ncols);
    float mean = csum[c] * (1.0f / SS);
    float var  = csq[c] * (1.0f / SS) - mean * mean;
    float sc   = g[c] * rsqrtf(var + BN_EPS);
    float y = (X[idx] - mean) * sc + b[c];
    out[idx] = f2bf(fmaxf(y, 0.f));
}

__global__ void k_extract(const float* __restrict__ T16, const float* __restrict__ bias,
                          float* __restrict__ out, int sigm) {
    int s = blockIdx.x * 256 + threadIdx.x;
    if (s >= SS) return;
    float v = T16[(size_t)s * 16] + bias[0];
    if (sigm) v = 1.0f / (1.0f + __expf(-v));
    out[s] = v;
}

// sp = pooled * gaus * sh0, written into zero-padded BEV bf16 buffer
__global__ void k_sp(const float* __restrict__ pooled, const float* __restrict__ sh0,
                     const float* __restrict__ gaus, u16* __restrict__ xpad) {
    int s = blockIdx.x, c = threadIdx.x;
    int b = s / (HH * WW), rem = s % (HH * WW), h = rem / WW, w = rem % WW;
    float v = pooled[(size_t)s * CC + c] * gaus[s] * sh0[s];
    xpad[(((size_t)(b * HP + h + 1)) * WP + (w + 1)) * CC + c] = f2bf(v);
}

// ---------------------------------------------------------------------------
// 3x3 conv, implicit GEMM with WMMA. Block = 8 waves covering 16 pixels x
// 128 output channels. Per-tap 16x256 bf16 activation tile staged in LDS,
// double-buffered with CDNA5 async global->LDS loads when available.
// ---------------------------------------------------------------------------
__global__ void k_conv(const u16* __restrict__ xpad, const u16* __restrict__ wT,
                       const float* __restrict__ bias, float* __restrict__ out) {
    __shared__ __align__(16) u16 tile[2][16 * CC];   // 2 x 8 KB
    int tid = threadIdx.x;
    int ocHalf = blockIdx.x, w0 = blockIdx.y * 16, bh = blockIdx.z;
    int b = bh / HH, h = bh % HH;
    int wave = tid >> 5, lane = tid & 31, half = lane >> 4, mn = lane & 15;
    int oc0 = ocHalf * 128 + wave * 16;
    const u32* bpBase = (const u32*)(wT + (size_t)(oc0 + mn) * 2304);
    __builtin_prefetch(bpBase, 0, 3);
    int tr = tid >> 4, tc = tid & 15;                // cooperative LDS fill mapping
    float bs = bias ? bias[oc0 + mn] : 0.f;
    v8f acc = {};

#if HAS_ASYNC_LDS
    // issue one tap's 8KB tile fill: 2 async b128 per lane, no VGPR round-trip
    auto issue = [&](int tap, int bufid) {
        int dy = tap / 3, dx = tap % 3;
        const u16* src = xpad + (((size_t)(b * HP + h + dy)) * WP + (w0 + tr + dx)) * CC
                         + tc * 16;
        u16* dst = &tile[bufid][tr * CC + tc * 16];
        __builtin_amdgcn_global_load_async_to_lds_b128(to_as1(src), to_as3(dst), 0, 0);
        __builtin_amdgcn_global_load_async_to_lds_b128(to_as1(src), to_as3(dst), 16, 0);
    };
    issue(0, 0);
    for (int tap = 0; tap < 9; tap++) {
        if (tap < 8) {
            issue(tap + 1, (tap + 1) & 1);            // prefetch next tap
            __builtin_amdgcn_s_wait_asynccnt(2);      // drain current tap, keep next in flight
        } else {
            __builtin_amdgcn_s_wait_asynccnt(0);
        }
        __syncthreads();
        const u32* lds = (const u32*)&tile[tap & 1][0];
        #pragma unroll
        for (int ks = 0; ks < 8; ks++) {
            int kc0 = ks * 32;
            ABFrag a, bf;
            int au = ((mn * CC + kc0) >> 1) + half * 8;
            #pragma unroll
            for (int j = 0; j < 8; j++) a.u[j] = lds[au + j];
            int bu = ((tap * CC + kc0) >> 1) + half * 8;
            #pragma unroll
            for (int j = 0; j < 8; j++) bf.u[j] = bpBase[bu + j];
            acc = __builtin_amdgcn_wmma_f32_16x16x32_bf16(false, a.v, false, bf.v,
                                                          (short)0, acc, false, false);
        }
        __syncthreads();   // buffer consumed; safe to refill two taps later
    }
#else
    for (int tap = 0; tap < 9; tap++) {
        int dy = tap / 3, dx = tap % 3;
        const uint4* src = (const uint4*)(xpad +
            (((size_t)(b * HP + h + dy)) * WP + (w0 + tr + dx)) * CC);
        uint4* dst = (uint4*)(&tile[0][tr * CC]);
        dst[tc * 2]     = src[tc * 2];
        dst[tc * 2 + 1] = src[tc * 2 + 1];
        __syncthreads();
        const u32* lds = (const u32*)&tile[0][0];
        #pragma unroll
        for (int ks = 0; ks < 8; ks++) {
            int kc0 = ks * 32;
            ABFrag a, bf;
            int au = ((mn * CC + kc0) >> 1) + half * 8;
            #pragma unroll
            for (int j = 0; j < 8; j++) a.u[j] = lds[au + j];
            int bu = ((tap * CC + kc0) >> 1) + half * 8;
            #pragma unroll
            for (int j = 0; j < 8; j++) bf.u[j] = bpBase[bu + j];
            acc = __builtin_amdgcn_wmma_f32_16x16x32_bf16(false, a.v, false, bf.v,
                                                          (short)0, acc, false, false);
        }
        __syncthreads();
    }
#endif

    int sBase = (b * HH + h) * WW + w0;
    #pragma unroll
    for (int r = 0; r < 8; r++) {
        int m = half * 8 + r;
        out[(size_t)(sBase + m) * CC + oc0 + mn] = acc[r] + bs;
    }
}

// BN+ReLU of conv1 output, re-packed into the padded bf16 BEV buffer
__global__ void k_bnrelu_pad(const float* __restrict__ X, const float* __restrict__ csum,
                             const float* __restrict__ csq, const float* __restrict__ g,
                             const float* __restrict__ bb, u16* __restrict__ xpad) {
    int s = blockIdx.x, c = threadIdx.x;
    float mean = csum[c] * (1.0f / SS);
    float var  = csq[c] * (1.0f / SS) - mean * mean;
    float sc   = g[c] * rsqrtf(var + BN_EPS);
    float y = (X[(size_t)s * CC + c] - mean) * sc + bb[c];
    int b = s / (HH * WW), rem = s % (HH * WW), h = rem / WW, w = rem % WW;
    xpad[(((size_t)(b * HP + h + 1)) * WP + (w + 1)) * CC + c] = f2bf(fmaxf(y, 0.f));
}

__global__ void k_bn_final(const float* __restrict__ X, const float* __restrict__ csum,
                           const float* __restrict__ csq, const float* __restrict__ g,
                           const float* __restrict__ bb, float* __restrict__ out) {
    size_t idx = (size_t)blockIdx.x * 256 + threadIdx.x;
    if (idx >= (size_t)SS * CC) return;
    int c = (int)(idx & 255);
    float mean = csum[c] * (1.0f / SS);
    float var  = csq[c] * (1.0f / SS) - mean * mean;
    float sc   = g[c] * rsqrtf(var + BN_EPS);
    out[idx] = fmaxf((X[idx] - mean) * sc + bb[c], 0.f);
}

// ---------------------------------------------------------------------------
extern "C" void kernel_launch(void* const* d_in, const int* in_sizes, int n_in,
                              void* d_out, int out_size, void* d_ws, size_t ws_size,
                              hipStream_t stream) {
    (void)in_sizes; (void)n_in; (void)out_size; (void)ws_size;
    const float* feats = (const float*)d_in[0];
    const int* cb = (const int*)d_in[1];
    const int* cy = (const int*)d_in[2];
    const int* cx = (const int*)d_in[3];
    const float* sh_w1 = (const float*)d_in[4];
    const float* sh_g1 = (const float*)d_in[5];
    const float* sh_b1 = (const float*)d_in[6];
    const float* sh_w2 = (const float*)d_in[7];
    const float* sh_g2 = (const float*)d_in[8];
    const float* sh_b2 = (const float*)d_in[9];
    const float* sh_w3 = (const float*)d_in[10];
    const float* sh_bias3 = (const float*)d_in[11];
    const float* g_w1 = (const float*)d_in[12];
    const float* g_g1 = (const float*)d_in[13];
    const float* g_b1 = (const float*)d_in[14];
    const float* g_w2 = (const float*)d_in[15];
    const float* g_g2 = (const float*)d_in[16];
    const float* g_b2 = (const float*)d_in[17];
    const float* g_w3 = (const float*)d_in[18];
    const float* g_bias3 = (const float*)d_in[19];
    const float* cw1 = (const float*)d_in[20];
    const float* c1_g = (const float*)d_in[21];
    const float* c1_b = (const float*)d_in[22];
    const float* cw2 = (const float*)d_in[23];
    const float* c2_bias = (const float*)d_in[24];
    const float* c2_g = (const float*)d_in[25];
    const float* c2_b = (const float*)d_in[26];

    char* ws = (char*)d_ws;
    size_t off = 0;
    auto take = [&](size_t bytes) -> char* {
        char* p = ws + off; off += (bytes + 255) & ~(size_t)255; return p;
    };
    float* sums     = (float*)take((size_t)SS * CC * 4);   // pooled f32 (in place)
    float* cnt      = (float*)take((size_t)SS * 4);
    float* t32      = (float*)take((size_t)SS * CC * 4);   // pre-BN temp; pw aliases here
    float* pw       = t32;                                 // [P,C] live only before GEMM1
    u16*   actA     = (u16*)take((size_t)SS * CC * 2);
    u16*   actB     = (u16*)take((size_t)SS * CC * 2);
    u16*   pooledbf = (u16*)take((size_t)SS * CC * 2);
    float* sh0      = (float*)take((size_t)SS * 4);
    float* gaus     = (float*)take((size_t)SS * 4);
    u16*   xpad     = (u16*)take((size_t)BB * HP * WP * CC * 2);
    u16*   shW1T    = (u16*)take(256 * 256 * 2);
    u16*   shW2T    = (u16*)take(256 * 256 * 2);
    u16*   shW3T    = (u16*)take(16 * 256 * 2);
    u16*   gW1T     = (u16*)take(256 * 256 * 2);
    u16*   gW2T     = (u16*)take(128 * 256 * 2);
    u16*   gW3T     = (u16*)take(16 * 128 * 2);
    u16*   cw1T     = (u16*)take((size_t)256 * 2304 * 2);
    u16*   cw2T     = (u16*)take((size_t)256 * 2304 * 2);
    float* csum     = (float*)take(512 * 4);
    float* csq      = csum + 256;

    auto gemm = [&](const u16* A, const u16* WT, float* D, int K, int Nout, int WN) {
        dim3 g(Nout / (16 * WN), SS / (16 * (8 / WN)));
        k_gemm<<<g, 256, 0, stream>>>(A, WT, D, K, Nout, WN);
    };
    auto stats = [&](const float* X, int ncols) {
        (void)hipMemsetAsync(csum, 0, 512 * 4, stream);
        k_colstats<<<SS / RPB, 256, 0, stream>>>(X, csum, csq, ncols);
    };

    // Stage 1-2: pooling
    (void)hipMemsetAsync(sums, 0, (size_t)SS * CC * 4, stream);
    (void)hipMemsetAsync(cnt, 0, (size_t)SS * 4, stream);
    k_transpose<<<dim3(NN / 32, CC / 32, BB), dim3(32, 8), 0, stream>>>(feats, pw);
    k_scatter<<<PP, 256, 0, stream>>>(pw, cb, cy, cx, sums, cnt);
    k_finalize<<<SS, 256, 0, stream>>>(sums, cnt, pooledbf);

    // Weight transposes to bf16 [Nout, K]
    k_wtrans<<<dim3(256, 1), 256, 0, stream>>>(sh_w1, shW1T, 256, 256);
    k_wtrans<<<dim3(256, 1), 256, 0, stream>>>(sh_w2, shW2T, 256, 256);
    k_wtrans<<<dim3(16, 1), 256, 0, stream>>>(sh_w3, shW3T, 256, 9);
    k_wtrans<<<dim3(256, 1), 256, 0, stream>>>(g_w1, gW1T, 256, 256);
    k_wtrans<<<dim3(128, 1), 256, 0, stream>>>(g_w2, gW2T, 256, 128);
    k_wtrans<<<dim3(16, 1), 256, 0, stream>>>(g_w3, gW3T, 128, 1);
    k_wtrans<<<dim3(256, 9), 256, 0, stream>>>(cw1, cw1T, 2304, 256);
    k_wtrans<<<dim3(256, 9), 256, 0, stream>>>(cw2, cw2T, 2304, 256);

    // SH head
    gemm(pooledbf, shW1T, t32, 256, 256, 4);
    stats(t32, 256);
    k_bnrelu_bf<<<SS, 256, 0, stream>>>(t32, csum, csq, sh_g1, sh_b1, actA, 256);
    gemm(actA, shW2T, t32, 256, 256, 4);
    stats(t32, 256);
    k_bnrelu_bf<<<SS, 256, 0, stream>>>(t32, csum, csq, sh_g2, sh_b2, actB, 256);
    gemm(actB, shW3T, t32, 256, 16, 1);
    k_extract<<<(SS + 255) / 256, 256, 0, stream>>>(t32, sh_bias3, sh0, 0);

    // Gaussian head
    gemm(pooledbf, gW1T, t32, 256, 256, 4);
    stats(t32, 256);
    k_bnrelu_bf<<<SS, 256, 0, stream>>>(t32, csum, csq, g_g1, g_b1, actA, 256);
    gemm(actA, gW2T, t32, 256, 128, 4);
    stats(t32, 128);
    k_bnrelu_bf<<<SS / 2, 256, 0, stream>>>(t32, csum, csq, g_g2, g_b2, actB, 128);
    gemm(actB, gW3T, t32, 128, 16, 1);
    k_extract<<<(SS + 255) / 256, 256, 0, stream>>>(t32, g_bias3, gaus, 1);

    // Modulated sparse features into padded BEV
    (void)hipMemsetAsync(xpad, 0, (size_t)BB * HP * WP * CC * 2, stream);
    k_sp<<<SS, 256, 0, stream>>>(sums, sh0, gaus, xpad);

    // conv_out: 3x3 no-bias + BN + ReLU
    k_conv<<<dim3(2, WW / 16, BB * HH), 256, 0, stream>>>(xpad, cw1T, nullptr, t32);
    stats(t32, 256);
    k_bnrelu_pad<<<SS, 256, 0, stream>>>(t32, csum, csq, c1_g, c1_b, xpad);

    // shared_conv: 3x3 + bias + BN + ReLU -> d_out (f32)
    k_conv<<<dim3(2, WW / 16, BB * HH), 256, 0, stream>>>(xpad, cw2T, c2_bias, t32);
    stats(t32, 256);
    k_bn_final<<<SS, 256, 0, stream>>>(t32, csum, csq, c2_g, c2_b, (float*)d_out);
}
